// SimpleBestRQ_13915694039131
// MI455X (gfx1250) — matrix operand order repaired and checked by reading, not compiled, not run on hardware
//
#include <hip/hip_runtime.h>
#include <hip/hip_fp16.h>
#include <stdint.h>
#include <stddef.h>

typedef __attribute__((ext_vector_type(16))) _Float16 v16h;
typedef __attribute__((ext_vector_type(8)))  _Float16 v8h;
typedef __attribute__((ext_vector_type(8)))  float    v8f;
typedef __attribute__((ext_vector_type(4)))  float    v4f;

#define DDIM   512
#define VCODES 8192
#define NROWS  32768
#define KSTEPS 16   // 512 / 32

// ---------------------------------------------------------------------------
// Kernel 1: transpose + convert projection to f16:  Pt[n][k] = (f16) P[k][n]
// ---------------------------------------------------------------------------
__global__ void k_proj_cvt(const float* __restrict__ P, _Float16* __restrict__ Pt) {
    int idx = blockIdx.x * blockDim.x + threadIdx.x;   // over 512*512
    int k = idx >> 9;
    int n = idx & 511;
    Pt[(size_t)n * DDIM + k] = (_Float16)P[(size_t)k * DDIM + n];
}

// ---------------------------------------------------------------------------
// Kernel 2: L2-normalize each codebook row, convert to f16 (row-major).
// ---------------------------------------------------------------------------
__global__ void k_codes_norm(const float* __restrict__ C, _Float16* __restrict__ Cn) {
    int wave = (int)((blockIdx.x * blockDim.x + threadIdx.x) >> 5);
    int lane = threadIdx.x & 31;
    if (wave >= VCODES) return;
    const float* row = C + (size_t)wave * DDIM;
    float vals[16];
    float ss = 0.0f;
#pragma unroll
    for (int i = 0; i < 16; ++i) {
        float v = row[lane + 32 * i];
        vals[i] = v;
        ss += v * v;
    }
#pragma unroll
    for (int off = 16; off > 0; off >>= 1)
        ss += __shfl_xor(ss, off, 32);
    float scale = 1.0f / fmaxf(sqrtf(ss), 1e-12f);
    _Float16* out = Cn + (size_t)wave * DDIM;
#pragma unroll
    for (int i = 0; i < 16; ++i)
        out[lane + 32 * i] = (_Float16)(vals[i] * scale);
}

// ---------------------------------------------------------------------------
// Kernel 3: projection GEMM  XP = f16( X @ P )   [32768 x 512]
// ---------------------------------------------------------------------------
__global__ void __launch_bounds__(256)
k_proj_gemm(const float* __restrict__ X, const _Float16* __restrict__ Pt,
            _Float16* __restrict__ XP) {
    int wave = threadIdx.x >> 5;
    int lane = threadIdx.x & 31;
    int half = lane >> 4;
    int l16  = lane & 15;
    int rowbase = (blockIdx.x * 8 + wave) * 16;

    const float* xrow = X + (size_t)(rowbase + l16) * DDIM;
    v16h a[KSTEPS];
#pragma unroll
    for (int s = 0; s < KSTEPS; ++s) {
        int kk = s * 32 + half * 16;
        v4f x0 = *(const v4f*)(xrow + kk);
        v4f x1 = *(const v4f*)(xrow + kk + 4);
        v4f x2 = *(const v4f*)(xrow + kk + 8);
        v4f x3 = *(const v4f*)(xrow + kk + 12);
        v16h t;
#pragma unroll
        for (int j = 0; j < 4; ++j) {
            t[j]      = (_Float16)x0[j];
            t[j + 4]  = (_Float16)x1[j];
            t[j + 8]  = (_Float16)x2[j];
            t[j + 12] = (_Float16)x3[j];
        }
        a[s] = t;
    }

#pragma unroll 1
    for (int nt = 0; nt < DDIM / 16; ++nt) {
        const _Float16* bcol = Pt + (size_t)(nt * 16 + l16) * DDIM + half * 16;
        v8f acc = {};
#pragma unroll
        for (int s = 0; s < KSTEPS; ++s) {
            v16h b = *(const v16h*)(bcol + s * 32);
            acc = __builtin_amdgcn_wmma_f32_16x16x32_f16(
                false, a[s], false, b, (short)0, acc, false, false);
        }
        _Float16* orow = XP + (size_t)(rowbase + half * 8) * DDIM + nt * 16 + l16;
#pragma unroll
        for (int r = 0; r < 8; ++r)
            orow[(size_t)r * DDIM] = (_Float16)acc[r];
    }
}

// ---------------------------------------------------------------------------
// Async 64-byte-per-thread tile fill: global -> LDS, ASYNCcnt-tracked.
// 256 threads move one 16-code x 512 f16 tile (16 KB, contiguous in Cn).
// ---------------------------------------------------------------------------
__device__ __forceinline__ void async_fill_tile(const _Float16* gbase,
                                                _Float16* lbase, int tid) {
    const _Float16* g = gbase + (size_t)tid * 32;            // 64 B per thread
    uint32_t l = (uint32_t)(uintptr_t)(lbase + tid * 32);    // LDS byte offset
#pragma unroll
    for (int i = 0; i < 4; ++i) {
        asm volatile("global_load_async_to_lds_b128 %0, %1, off"
                     :: "v"(l + i * 16), "v"(g + i * 8)
                     : "memory");
    }
}

// ---------------------------------------------------------------------------
// Kernel 4: fused score GEMM + argmax with double-buffered LDS code tiles.
// Block = 8 waves x 16 rows = 128 rows sharing each code tile (8x L2 cut).
// Two independent WMMA accumulator chains (even/odd k) for XDL ILP.
// ---------------------------------------------------------------------------
__global__ void __launch_bounds__(256)
k_argmax(const _Float16* __restrict__ XP, const _Float16* __restrict__ Cn,
         const uint8_t* __restrict__ mask, int* __restrict__ out) {
    __shared__ _Float16 s_b[2][16][DDIM];   // 2 x 16 KB code tiles
    __shared__ float s_sc[8][256];          // 8 KB
    __shared__ int   s_id[8][256];          // 8 KB

    int tid  = threadIdx.x;
    int wave = tid >> 5;
    int lane = tid & 31;
    int half = lane >> 4;
    int l16  = lane & 15;
    int rowbase = (blockIdx.x * 8 + wave) * 16;

    // A fragments: 16 rows of XP, register-resident across all 512 code tiles
    const _Float16* xrow = XP + (size_t)(rowbase + l16) * DDIM;
    v16h a[KSTEPS];
#pragma unroll
    for (int s = 0; s < KSTEPS; ++s) {
        int kk = s * 32 + half * 16;
        v8h lo = *(const v8h*)(xrow + kk);
        v8h hi = *(const v8h*)(xrow + kk + 8);
        v16h t;
        ((v8h*)&t)[0] = lo;
        ((v8h*)&t)[1] = hi;
        a[s] = t;
    }

    float best_s[8];
    int   best_i[8];
#pragma unroll
    for (int r = 0; r < 8; ++r) { best_s[r] = -3.4e38f; best_i[r] = 0; }

    // Prologue: fill tile 0 into buffer 0
    async_fill_tile(Cn, &s_b[0][0][0], tid);
    asm volatile("s_wait_asynccnt 0x0" ::: "memory");
    __syncthreads();

#pragma unroll 1
    for (int ct = 0; ct < VCODES / 16; ++ct) {
        int buf = ct & 1;
        if (ct + 1 < VCODES / 16)
            async_fill_tile(Cn + (size_t)(ct + 1) * 16 * DDIM,
                            &s_b[buf ^ 1][0][0], tid);

        const _Float16* bbase = &s_b[buf][l16][half * 16];
        v8f acc0 = {};
        v8f acc1 = {};
#pragma unroll
        for (int s = 0; s < KSTEPS; s += 2) {
            v16h b0 = *(const v16h*)(bbase + s * 32);
            v16h b1 = *(const v16h*)(bbase + s * 32 + 32);
            acc0 = __builtin_amdgcn_wmma_f32_16x16x32_f16(
                false, a[s],     false, b0, (short)0, acc0, false, false);
            acc1 = __builtin_amdgcn_wmma_f32_16x16x32_f16(
                false, a[s + 1], false, b1, (short)0, acc1, false, false);
        }

        int code = ct * 16 + l16;
#pragma unroll
        for (int r = 0; r < 8; ++r) {
            float sc = acc0[r] + acc1[r];
            if (sc > best_s[r]) { best_s[r] = sc; best_i[r] = code; }
        }

        // next tile's fill has had the whole compute phase to land
        asm volatile("s_wait_asynccnt 0x0" ::: "memory");
        __syncthreads();
    }

#pragma unroll
    for (int r = 0; r < 8; ++r) {
        int slot = (r + 8 * half) * 16 + l16;   // [row][candidate]
        s_sc[wave][slot] = best_s[r];
        s_id[wave][slot] = best_i[r];
    }
    __syncthreads();

    if (lane < 16) {
        float bs = -3.4e38f;
        int   bi = 0x7fffffff;
#pragma unroll
        for (int j = 0; j < 16; ++j) {
            float sc = s_sc[wave][lane * 16 + j];
            int   ci = s_id[wave][lane * 16 + j];
            if (sc > bs || (sc == bs && ci < bi)) { bs = sc; bi = ci; }
        }
        int gid = rowbase + lane;
        out[gid] = mask[gid] ? bi : 0;
    }
}

// ---------------------------------------------------------------------------
extern "C" void kernel_launch(void* const* d_in, const int* in_sizes, int n_in,
                              void* d_out, int out_size, void* d_ws, size_t ws_size,
                              hipStream_t stream) {
    const float*   x     = (const float*)d_in[0];      // [8,4096,512]
    const uint8_t* mask  = (const uint8_t*)d_in[1];    // [8,4096] bool
    const float*   proj  = (const float*)d_in[2];      // [512,512]
    const float*   codes = (const float*)d_in[3];      // [8192,512]
    int*           out   = (int*)d_out;                // [8,4096] int32

    char* ws = (char*)d_ws;
    _Float16* Pt = (_Float16*)ws;                                       // 0.5 MB
    _Float16* Cn = (_Float16*)(ws + (size_t)DDIM * DDIM * 2);           // 8 MB
    _Float16* XP = (_Float16*)(ws + (size_t)DDIM * DDIM * 2
                                  + (size_t)VCODES * DDIM * 2);         // 33.5 MB

    k_proj_cvt  <<<(DDIM * DDIM) / 256, 256, 0, stream>>>(proj, Pt);
    k_codes_norm<<<VCODES / 8,          256, 0, stream>>>(codes, Cn);
    k_proj_gemm <<<NROWS / 128,         256, 0, stream>>>(x, Pt, XP);
    k_argmax    <<<NROWS / 128,         256, 0, stream>>>(XP, Cn, mask, out);
}
